// stableKT_11553462026797
// MI455X (gfx1250) — matrix operand order repaired and checked by reading, not compiled, not run on hardware
//
#include <hip/hip_runtime.h>
#include <hip/hip_bf16.h>
#include <math.h>

// ---------------- problem constants ----------------
constexpr int BS   = 16;
constexpr int S    = 512;
constexpr int D    = 1024;
constexpr int H    = 16;
constexpr int DK   = 64;
constexpr int HALF = 8;
constexpr int M    = BS * S;      // 8192 rows

typedef __attribute__((ext_vector_type(16))) __bf16   v16bf;
typedef __attribute__((ext_vector_type(8)))  float    v8f;
typedef __attribute__((ext_vector_type(4)))  unsigned v4u;
typedef __attribute__((ext_vector_type(8)))  int      v8i;
typedef __attribute__((ext_vector_type(4)))  int      v4i;

union U16 { v16bf v; uint4 q[2]; };

// float -> bf16, round-to-nearest-even (pure bit math)
__device__ inline __bf16 f2bf(float x) {
    union { float f; unsigned u; } a; a.f = x;
    unsigned r = a.u + 0x7FFFu + ((a.u >> 16) & 1u);
    union { unsigned short s; __bf16 b; } c;
    c.s = (unsigned short)(r >> 16);
    return c.b;
}

__device__ inline v8f wmma_bf16(v16bf a, v16bf b, v8f c) {
    // D(16x16,f32) = A(16x32,bf16) * B(32x16,bf16) + C
    return __builtin_amdgcn_wmma_f32_16x16x32_bf16(false, a, false, b,
                                                   (short)0, c, false, false);
}

// ---------------- Tensor Data Mover: 2D tile -> LDS ----------------
// Builds D# per CDNA5 ISA ch.8 (group0: count/lds/global/type, group1: dims).
// data_size = 2 bytes (bf16). Row-major packing into LDS.
// This toolchain exposes the 6-arg builtin: (g0, g1, g2, g3, g_extra, cpol).
__device__ inline void tdm_load_2d(const void* lds_dst, const void* gsrc,
                                   unsigned tensor_d0, unsigned tensor_d1,
                                   unsigned tile_d0, unsigned tile_d1,
                                   unsigned stride0_elems)
{
    unsigned long long ga = (unsigned long long)(uintptr_t)gsrc;
    unsigned lds_off = (unsigned)(uintptr_t)lds_dst;   // low 32 bits of generic = LDS byte addr
    v4u g0;
    g0[0] = 1u;                                        // count=1, user descriptor
    g0[1] = lds_off;                                   // lds_addr (bytes)
    g0[2] = (unsigned)(ga & 0xFFFFFFFFu);              // global_addr[31:0]
    g0[3] = (unsigned)((ga >> 32) & 0x01FFFFFFu) | (2u << 30);  // [56:32] + type=2
    v8i g1;
    g1[0] = (int)(1u << 16);                           // wg_mask=0, data_size=1 (2B)
    g1[1] = (int)((tensor_d0 & 0xFFFFu) << 16);        // tensor_dim0[15:0] @ bits63:48
    g1[2] = (int)(((tensor_d0 >> 16) & 0xFFFFu) |      // tensor_dim0[31:16]
                  ((tensor_d1 & 0xFFFFu) << 16));      // tensor_dim1[15:0]
    g1[3] = (int)(((tensor_d1 >> 16) & 0xFFFFu) |      // tensor_dim1[31:16]
                  ((tile_d0 & 0xFFFFu) << 16));        // tile_dim0
    g1[4] = (int)(tile_d1 & 0xFFFFu);                  // tile_dim1; tile_dim2=0 (2D)
    g1[5] = (int)stride0_elems;                        // tensor_dim0_stride[31:0]
    g1[6] = 0;                                         // stride0[47:32], stride1[15:0]
    g1[7] = 0;                                         // stride1[47:16]
    v4i z4 = {0, 0, 0, 0};                             // groups 2/3 unused for 2D
    v8i z8 = {0, 0, 0, 0, 0, 0, 0, 0};                 // extra group (unused)
    __builtin_amdgcn_tensor_load_to_lds(g0, g1, z4, z4, z8, 0);
}

// A-operand tile (16x32): lane l holds row m=l&15, K runs {k0..k0+7, k0+16..k0+23}, k0=(l>>4)*8
__device__ inline v16bf load_a_g(const __bf16* p, int ld, int row0, int kk) {
    int l = threadIdx.x & 31;
    const __bf16* r = p + (size_t)(row0 + (l & 15)) * ld + kk + ((l >> 4) * 8);
    U16 u;
    u.q[0] = *(const uint4*)(r);
    u.q[1] = *(const uint4*)(r + 16);
    return u.v;
}

// B-operand tile (32x16): column n=l&15 == row n of B^T (row-major), K runs {k0..k0+7, k0+16..k0+23}, k0=(l>>4)*8
__device__ inline v16bf load_b_any(const __bf16* p, int ld, int n0, int kk) {
    int l = threadIdx.x & 31;
    const __bf16* r = p + (size_t)(n0 + (l & 15)) * ld + kk + ((l >> 4) * 8);
    U16 u;
    u.q[0] = *(const uint4*)(r);
    u.q[1] = *(const uint4*)(r + 16);
    return u.v;
}

// A-operand tile from per-wave LDS scratch laid out row-major [16][32]
__device__ inline v16bf load_a_lds(const __bf16* sm) {
    int l = threadIdx.x & 31;
    const __bf16* r = sm + (l & 15) * 32 + ((l >> 4) * 8);
    U16 u;
    u.q[0] = *(const uint4*)(r);
    u.q[1] = *(const uint4*)(r + 16);
    return u.v;
}

// reductions within 16-lane halves (wave32: xor masks keep bit4)
__device__ inline float rmax16(float v) {
    v = fmaxf(v, __shfl_xor(v, 1));
    v = fmaxf(v, __shfl_xor(v, 2));
    v = fmaxf(v, __shfl_xor(v, 4));
    v = fmaxf(v, __shfl_xor(v, 8));
    return v;
}
__device__ inline float rsum16(float v) {
    v += __shfl_xor(v, 1);
    v += __shfl_xor(v, 2);
    v += __shfl_xor(v, 4);
    v += __shfl_xor(v, 8);
    return v;
}
__device__ inline float rsum32(float v) {
    v += __shfl_xor(v, 16);
    return rsum16(v);
}

// ---------------- kernel 1: fp32 -> bf16 convert ----------------
__global__ void cvt_kernel(const float* __restrict__ in, __bf16* __restrict__ out, int n) {
    int i = blockIdx.x * 256 + threadIdx.x;
    if (i < n) out[i] = f2bf(in[i]);
}

// ---------------- kernel 2: QKV projection GEMM ----------------
// 4 waves/block; 64x1024 weight slab TDM-staged in LDS, each wave owns 16 rows of A.
__global__ __launch_bounds__(128) void proj_kernel(
    const __bf16* __restrict__ xh,
    const __bf16* __restrict__ Wqh, const __bf16* __restrict__ Wkh, const __bf16* __restrict__ Wvh,
    const float* __restrict__ bq, const float* __restrict__ bk, const float* __restrict__ bv,
    float* __restrict__ Qf, float* __restrict__ Kf, float* __restrict__ Vf)
{
    extern __shared__ __align__(16) char smem[];
    __bf16* smW = (__bf16*)smem;                        // [64][1024] bf16 = 128KB

    const __bf16* W; const float* bias; float* out;
    if (blockIdx.z == 0)      { W = Wqh; bias = bq; out = Qf; }
    else if (blockIdx.z == 1) { W = Wkh; bias = bk; out = Kf; }
    else                      { W = Wvh; bias = bv; out = Vf; }

    const int wid  = threadIdx.x >> 5;
    const int h    = blockIdx.y;                        // 64-wide N block == one head
    const int row0 = blockIdx.x * 64 + wid * 16;
    const int lane = threadIdx.x & 31;
    const int hf   = lane >> 4;
    const int lidx = lane & 15;

    if (wid == 0) {   // one wave drives the TDM (EXEC ignored; wave-level DMA)
        tdm_load_2d(smW, W + (size_t)h * 64 * D,
                    /*tensor_d0=*/D, /*tensor_d1=*/64,
                    /*tile_d0=*/D, /*tile_d1=*/64, /*stride0=*/D);
        __builtin_amdgcn_s_wait_tensorcnt((short)0);
    }
    __syncthreads();

    v8f a0 = {}, a1 = {}, a2 = {}, a3 = {};
    for (int kk = 0; kk < D; kk += 32) {
        if (kk + 64 < D)
            __builtin_prefetch(xh + (size_t)(row0 + lidx) * D + kk + 64, 0, 3);
        v16bf a  = load_a_g(xh, D, row0, kk);
        v16bf b0 = load_b_any(smW, D,  0, kk);
        v16bf b1 = load_b_any(smW, D, 16, kk);
        v16bf b2 = load_b_any(smW, D, 32, kk);
        v16bf b3 = load_b_any(smW, D, 48, kk);
        a0 = wmma_bf16(a, b0, a0);
        a1 = wmma_bf16(a, b1, a1);
        a2 = wmma_bf16(a, b2, a2);
        a3 = wmma_bf16(a, b3, a3);
    }
    const float bb0 = bias[h * 64 +  0 + lidx];
    const float bb1 = bias[h * 64 + 16 + lidx];
    const float bb2 = bias[h * 64 + 32 + lidx];
    const float bb3 = bias[h * 64 + 48 + lidx];
#pragma unroll
    for (int r = 0; r < 8; ++r) {
        const int mrow = row0 + r + 8 * hf;
        const int b = mrow >> 9, s = mrow & 511;
        float* orow = out + (((size_t)b * H + h) * S + s) * DK;
        orow[ 0 + lidx] = a0[r] + bb0;
        orow[16 + lidx] = a1[r] + bb1;
        orow[32 + lidx] = a2[r] + bb2;
        orow[48 + lidx] = a3[r] + bb3;
    }
}

// ---------------- kernel 3: head prep ----------------
__global__ __launch_bounds__(128) void prep_kernel(
    const float* __restrict__ Qf, const float* __restrict__ Kf, const float* __restrict__ Vf,
    __bf16* __restrict__ Qh, __bf16* __restrict__ Kh, __bf16* __restrict__ Vt,
    float* __restrict__ qn, float* __restrict__ qy, float* __restrict__ xqy,
    float* __restrict__ kn, float* __restrict__ ky, float* __restrict__ xky)
{
    const int row  = blockIdx.x * 4 + (threadIdx.x >> 5);   // (b*H+h)*S + s
    const int lane = threadIdx.x & 31;
    const int s  = row & (S - 1);
    const int bh = row >> 9;
    const int h  = bh & (H - 1);

    const float* q = Qf + (size_t)row * DK;
    const float* k = Kf + (size_t)row * DK;
    const float* v = Vf + (size_t)row * DK;

    __bf16* vt = Vt + (size_t)bh * DK * S;
    vt[(size_t)lane * S + s]        = f2bf(v[lane]);
    vt[(size_t)(lane + 32) * S + s] = f2bf(v[lane + 32]);

    __bf16* qo = Qh + (size_t)row * DK;
    __bf16* ko = Kh + (size_t)row * DK;

    if (h < HALF) {
        qo[lane] = f2bf(q[lane]); qo[lane + 32] = f2bf(q[lane + 32]);
        ko[lane] = f2bf(k[lane]); ko[lane + 32] = f2bf(k[lane + 32]);
    } else {
        {
            const float last = q[63];
            const float sig  = 1.0f / (1.0f + __expf(-last));
            const float x0 = q[lane] * sig;
            const float x1 = (lane + 32 == 63) ? 0.0f : q[lane + 32] * sig;
            qo[lane] = f2bf(x0); qo[lane + 32] = f2bf(x1);
            float ss = rsum32(x0 * x0 + x1 * x1);
            if (lane == 0) {
                qn[row]  = ss;
                qy[row]  = sig;
                xqy[row] = sqrtf(1.0f - sig * sig + 1e-6f);
            }
        }
        {
            const float last = k[63];
            const float sig  = 1.0f / (1.0f + __expf(-last));
            const float x0 = k[lane] * sig;
            const float x1 = (lane + 32 == 63) ? 0.0f : k[lane + 32] * sig;
            ko[lane] = f2bf(x0); ko[lane + 32] = f2bf(x1);
            float ss = rsum32(x0 * x0 + x1 * x1);
            if (lane == 0) {
                kn[row]  = ss;
                ky[row]  = sig;
                xky[row] = sqrtf(1.0f - sig * sig + 1e-6f);
            }
        }
    }
}

// penumbral score (R=1, gamma=0.1)
__device__ inline float pen_score(float dot, float qn_, float qy_, float xq_,
                                  float kn_, float ky_, float xk_)
{
    const float d2 = qn_ + kn_ - 2.0f * dot;
    const float pd = sqrtf(fmaxf(d2, 0.0f) + 1e-12f);
    const float qy2 = qy_ * qy_, ky2 = ky_ * ky_;
    const float t   = 0.5f * (xq_ + xk_ - pd);
    const float lca = fmaxf(fmaxf(qy2, ky2), 1.0f - t * t);
    const float u   = (pd * pd + ky2 - qy2) / (2.0f * pd + 1e-6f);
    const float lca_out = u * u + qy2;
    const float dpx = pd - xq_;
    const bool exists = (pd <= xq_) || (dpx * dpx + ky2 <= 1.0f);
    return -0.1f * (exists ? lca : lca_out);
}

// ---------------- kernel 4: fused flash attention ----------------
// One 8-wave block per (b,h). K (512x64) and V^T (64x512) slabs TDM-staged into
// LDS once; each wave owns query tiles qt = wid, wid+8, ... with online softmax.
// No __syncthreads in the per-wave loop (divergent trip counts); the per-wave
// P bounce relies on in-order DS within a wave.
__global__ __launch_bounds__(256) void attn_kernel(
    const __bf16* __restrict__ Qh, const __bf16* __restrict__ Kh, const __bf16* __restrict__ Vt,
    const float* __restrict__ qn, const float* __restrict__ qy, const float* __restrict__ xqy,
    const float* __restrict__ kn, const float* __restrict__ ky, const float* __restrict__ xky,
    __bf16* __restrict__ attnout)
{
    extern __shared__ __align__(16) char smem[];
    __bf16* smK = (__bf16*)smem;                          // [512][64]  = 64KB
    __bf16* smV = (__bf16*)(smem + (size_t)S * DK * 2);   // [64][512]  = 64KB

    const int h = blockIdx.x, b = blockIdx.y;
    const int wid  = threadIdx.x >> 5;
    const int lane = threadIdx.x & 31;
    const int hf = lane >> 4, lidx = lane & 15;
    const int bh = b * H + h;
    const bool pen = (h >= HALF);

    __bf16* smP = (__bf16*)(smem + 2 * (size_t)S * DK * 2) + wid * 512;  // 16x32/wave

    const __bf16* Qbh = Qh + (size_t)bh * S * DK;

    if (wid == 0) {
        tdm_load_2d(smK, Kh + (size_t)bh * S * DK,
                    /*tensor_d0=*/DK, /*tensor_d1=*/S,
                    /*tile_d0=*/DK, /*tile_d1=*/S, /*stride0=*/DK);
        tdm_load_2d(smV, Vt + (size_t)bh * DK * S,
                    /*tensor_d0=*/S, /*tensor_d1=*/DK,
                    /*tile_d0=*/S, /*tile_d1=*/DK, /*stride0=*/S);
        __builtin_amdgcn_s_wait_tensorcnt((short)0);
    }
    __syncthreads();

    const float slope = -exp2f(-0.5f * (float)(h + 1));   // analytic ALiBi slope
    const float inv   = 0.125f;                            // 1/sqrt(64)

    for (int qt = wid; qt < S / 16; qt += 8) {
        const int qbase = qt * 16;

        const v16bf aq0 = load_a_g(Qbh, DK, qbase, 0);
        const v16bf aq1 = load_a_g(Qbh, DK, qbase, 32);

        float qn8[8], qy8[8], xq8[8];
        if (pen) {
            const float* qnB = qn  + (size_t)bh * S + qbase;
            const float* qyB = qy  + (size_t)bh * S + qbase;
            const float* xqB = xqy + (size_t)bh * S + qbase;
#pragma unroll
            for (int r = 0; r < 8; ++r) {
                const int m = r + 8 * hf;
                qn8[r] = qnB[m]; qy8[r] = qyB[m]; xq8[r] = xqB[m];
            }
        }

        float mrun[8], lrun[8];
        v8f o0 = {}, o1 = {}, o2 = {}, o3 = {};
#pragma unroll
        for (int r = 0; r < 8; ++r) { mrun[r] = -1e30f; lrun[r] = 0.0f; }

        const int ktmax = qt >> 1;   // causal skip: only tiles with some j < i
        for (int kt = 0; kt <= ktmax; ++kt) {
            const int kbase = kt * 32;

            v8f s0 = {}, s1 = {};
            {
                v16bf b00 = load_b_any(smK, DK, kbase,      0);
                v16bf b01 = load_b_any(smK, DK, kbase,      32);
                v16bf b10 = load_b_any(smK, DK, kbase + 16, 0);
                v16bf b11 = load_b_any(smK, DK, kbase + 16, 32);
                s0 = wmma_bf16(aq0, b00, s0);
                s0 = wmma_bf16(aq1, b01, s0);
                s1 = wmma_bf16(aq0, b10, s1);
                s1 = wmma_bf16(aq1, b11, s1);
            }

            const int j0 = kbase + lidx, j1 = j0 + 16;
            float kn0 = 0, kn1 = 0, ky0 = 0, ky1 = 0, xk0 = 0, xk1 = 0;
            if (pen) {
                const float* knB = kn  + (size_t)bh * S;
                const float* kyB = ky  + (size_t)bh * S;
                const float* xkB = xky + (size_t)bh * S;
                kn0 = knB[j0]; kn1 = knB[j1];
                ky0 = kyB[j0]; ky1 = kyB[j1];
                xk0 = xkB[j0]; xk1 = xkB[j1];
            }

            float sc0[8], sc1[8];
#pragma unroll
            for (int r = 0; r < 8; ++r) {
                const int i = qbase + r + 8 * hf;
                float v0, v1;
                if (!pen) {
                    v0 = s0[r] * inv;
                    v1 = s1[r] * inv;
                } else {
                    v0 = pen_score(s0[r], qn8[r], qy8[r], xq8[r], kn0, ky0, xk0) * inv;
                    v1 = pen_score(s1[r], qn8[r], qy8[r], xq8[r], kn1, ky1, xk1) * inv;
                }
                v0 += slope * fabsf((float)(i - j0));
                v1 += slope * fabsf((float)(i - j1));
                if (j0 >= i) v0 = -1e30f;      // strict causal: allowed only j < i
                if (j1 >= i) v1 = -1e30f;
                sc0[r] = v0; sc1[r] = v1;
            }

            float p0[8], p1[8], scl[8];
#pragma unroll
            for (int r = 0; r < 8; ++r) {
                float rmax = rmax16(fmaxf(sc0[r], sc1[r]));
                const float mn = fmaxf(mrun[r], rmax);
                scl[r] = __expf(mrun[r] - mn);
                p0[r] = __expf(sc0[r] - mn);
                p1[r] = __expf(sc1[r] - mn);
                const float rs = rsum16(p0[r] + p1[r]);
                lrun[r] = lrun[r] * scl[r] + rs;
                mrun[r] = mn;
            }
#pragma unroll
            for (int r = 0; r < 8; ++r) {
                o0[r] *= scl[r]; o1[r] *= scl[r]; o2[r] *= scl[r]; o3[r] *= scl[r];
            }

            // P: C layout -> per-wave LDS -> A layout (DS in-order within wave)
#pragma unroll
            for (int r = 0; r < 8; ++r) {
                const int m = r + 8 * hf;
                smP[m * 32 + lidx]      = f2bf(p0[r]);
                smP[m * 32 + 16 + lidx] = f2bf(p1[r]);
            }
            const v16bf pa = load_a_lds(smP);

            v16bf vb0 = load_b_any(smV, S,  0, kbase);
            v16bf vb1 = load_b_any(smV, S, 16, kbase);
            v16bf vb2 = load_b_any(smV, S, 32, kbase);
            v16bf vb3 = load_b_any(smV, S, 48, kbase);
            o0 = wmma_bf16(pa, vb0, o0);
            o1 = wmma_bf16(pa, vb1, o1);
            o2 = wmma_bf16(pa, vb2, o2);
            o3 = wmma_bf16(pa, vb3, o3);
        }

        // normalize, zero query row 0 (reference: p[:,:,0,:] = 0), write bf16
#pragma unroll
        for (int r = 0; r < 8; ++r) {
            const int i = qbase + r + 8 * hf;
            float rn = (lrun[r] > 0.0f) ? (1.0f / lrun[r]) : 0.0f;
            if (i == 0) rn = 0.0f;
            __bf16* orow = attnout + ((size_t)(b * S + i)) * D + h * DK;
            orow[ 0 + lidx] = f2bf(o0[r] * rn);
            orow[16 + lidx] = f2bf(o1[r] * rn);
            orow[32 + lidx] = f2bf(o2[r] * rn);
            orow[48 + lidx] = f2bf(o3[r] * rn);
        }
    }
}

// ---------------- kernel 5: output projection GEMM ----------------
__global__ __launch_bounds__(128) void outproj_kernel(
    const __bf16* __restrict__ Ah, const __bf16* __restrict__ Woh,
    const float* __restrict__ bo, float* __restrict__ out)
{
    extern __shared__ __align__(16) char smem[];
    __bf16* smW = (__bf16*)smem;                        // [64][1024] bf16 = 128KB

    const int wid  = threadIdx.x >> 5;
    const int nb   = blockIdx.y * 64;
    const int row0 = blockIdx.x * 64 + wid * 16;
    const int lane = threadIdx.x & 31;
    const int hf = lane >> 4, lidx = lane & 15;

    if (wid == 0) {
        tdm_load_2d(smW, Woh + (size_t)nb * D,
                    /*tensor_d0=*/D, /*tensor_d1=*/64,
                    /*tile_d0=*/D, /*tile_d1=*/64, /*stride0=*/D);
        __builtin_amdgcn_s_wait_tensorcnt((short)0);
    }
    __syncthreads();

    v8f a0 = {}, a1 = {}, a2 = {}, a3 = {};
    for (int kk = 0; kk < D; kk += 32) {
        if (kk + 64 < D)
            __builtin_prefetch(Ah + (size_t)(row0 + lidx) * D + kk + 64, 0, 3);
        v16bf a  = load_a_g(Ah, D, row0, kk);
        v16bf b0 = load_b_any(smW, D,  0, kk);
        v16bf b1 = load_b_any(smW, D, 16, kk);
        v16bf b2 = load_b_any(smW, D, 32, kk);
        v16bf b3 = load_b_any(smW, D, 48, kk);
        a0 = wmma_bf16(a, b0, a0);
        a1 = wmma_bf16(a, b1, a1);
        a2 = wmma_bf16(a, b2, a2);
        a3 = wmma_bf16(a, b3, a3);
    }
    const float bb0 = bo[nb +  0 + lidx];
    const float bb1 = bo[nb + 16 + lidx];
    const float bb2 = bo[nb + 32 + lidx];
    const float bb3 = bo[nb + 48 + lidx];
#pragma unroll
    for (int r = 0; r < 8; ++r) {
        const int mrow = row0 + r + 8 * hf;
        float* orow = out + (size_t)mrow * D + nb;
        orow[ 0 + lidx] = a0[r] + bb0;
        orow[16 + lidx] = a1[r] + bb1;
        orow[32 + lidx] = a2[r] + bb2;
        orow[48 + lidx] = a3[r] + bb3;
    }
}

// ---------------- host orchestration ----------------
extern "C" void kernel_launch(void* const* d_in, const int* in_sizes, int n_in,
                              void* d_out, int out_size, void* d_ws, size_t ws_size,
                              hipStream_t stream) {
    (void)in_sizes; (void)n_in; (void)out_size; (void)ws_size;

    const float* x  = (const float*)d_in[0];
    const float* Wq = (const float*)d_in[1];
    const float* bq = (const float*)d_in[2];
    const float* Wk = (const float*)d_in[3];
    const float* bk = (const float*)d_in[4];
    const float* Wv = (const float*)d_in[5];
    const float* bv = (const float*)d_in[6];
    const float* Wo = (const float*)d_in[7];
    const float* bo = (const float*)d_in[8];
    // d_in[9] = alibi (computed analytically), d_in[10] = mask (analytic causal)
    float* out = (float*)d_out;

    char* w = (char*)d_ws;
    size_t off = 0;
    auto alloc = [&](size_t bytes) -> void* {
        void* p = w + off;
        off = (off + bytes + 255) & ~(size_t)255;
        return p;
    };

    const size_t nX = (size_t)M * D;     // 8388608
    const size_t nW = (size_t)D * D;     // 1048576
    const size_t nSide = (size_t)BS * H * S;

    __bf16* xh  = (__bf16*)alloc(nX * 2);
    __bf16* Wqh = (__bf16*)alloc(nW * 2);
    __bf16* Wkh = (__bf16*)alloc(nW * 2);
    __bf16* Wvh = (__bf16*)alloc(nW * 2);
    __bf16* Woh = (__bf16*)alloc(nW * 2);
    float*  Qf  = (float*)alloc(nX * 4);
    float*  Kf  = (float*)alloc(nX * 4);
    float*  Vf  = (float*)alloc(nX * 4);
    __bf16* Qh  = (__bf16*)alloc(nX * 2);
    __bf16* Kh  = (__bf16*)alloc(nX * 2);
    __bf16* Vt  = (__bf16*)alloc(nX * 2);
    float*  qn_  = (float*)alloc(nSide * 4);
    float*  qy_  = (float*)alloc(nSide * 4);
    float*  xqy_ = (float*)alloc(nSide * 4);
    float*  kn_  = (float*)alloc(nSide * 4);
    float*  ky_  = (float*)alloc(nSide * 4);
    float*  xky_ = (float*)alloc(nSide * 4);
    __bf16* ao  = (__bf16*)alloc(nX * 2);

    // 1) convert x + weights to bf16
    cvt_kernel<<<dim3((unsigned)((nX + 255) / 256)), 256, 0, stream>>>(x,  xh,  (int)nX);
    cvt_kernel<<<dim3((unsigned)((nW + 255) / 256)), 256, 0, stream>>>(Wq, Wqh, (int)nW);
    cvt_kernel<<<dim3((unsigned)((nW + 255) / 256)), 256, 0, stream>>>(Wk, Wkh, (int)nW);
    cvt_kernel<<<dim3((unsigned)((nW + 255) / 256)), 256, 0, stream>>>(Wv, Wvh, (int)nW);
    cvt_kernel<<<dim3((unsigned)((nW + 255) / 256)), 256, 0, stream>>>(Wo, Woh, (int)nW);

    const size_t GEMM_LDS = (size_t)64 * D * 2;                 // 128KB weight slab
    const size_t ATTN_LDS = 2 * (size_t)S * DK * 2 + 8 * 1024;  // K + V^T + P scratch

    // 2) QKV projection (WMMA, TDM-staged weights)
    proj_kernel<<<dim3(M / 64, H, 3), 128, GEMM_LDS, stream>>>(
        xh, Wqh, Wkh, Wvh, bq, bk, bv, Qf, Kf, Vf);

    // 3) head prep: bf16 forms, V transpose, penumbral scalars
    prep_kernel<<<dim3((unsigned)(nSide / 4)), 128, 0, stream>>>(
        Qf, Kf, Vf, Qh, Kh, Vt, qn_, qy_, xqy_, kn_, ky_, xky_);

    // 4) fused flash attention (TDM-staged K/V in LDS, WMMA scores + PV)
    attn_kernel<<<dim3(H, BS), 256, ATTN_LDS, stream>>>(
        Qh, Kh, Vt, qn_, qy_, xqy_, kn_, ky_, xky_, ao);

    // 5) output projection (WMMA, TDM-staged weights) -> f32 d_out
    outproj_kernel<<<dim3(M / 64, D / 64), 128, GEMM_LDS, stream>>>(ao, Woh, bo, out);
}
